// SparseAttention_35175782154389
// MI455X (gfx1250) — matrix operand order repaired and checked by reading, not compiled, not run on hardware
//
#include <hip/hip_runtime.h>

#define D_MODEL 256
#define N_BATCH 4
#define LQ      2048
#define LIN     4096
#define KTOP    409   // int(0.1 * 4096)

typedef __attribute__((ext_vector_type(16))) __bf16       v16bf;
typedef __attribute__((ext_vector_type(8)))  float        v8f;
typedef __attribute__((ext_vector_type(4)))  float        v4f;
typedef __attribute__((ext_vector_type(4)))  unsigned int v4u;

union Frag16 { v16bf v; v4u q[2]; };

static __device__ inline v8f wmma_bf16(v16bf a, v16bf b, v8f c) {
  // D = A(16x32 bf16) * B(32x16 bf16) + C(16x16 f32)
  return __builtin_amdgcn_wmma_f32_16x16x32_bf16(false, a, false, b, (short)0, c,
                                                 false, false);
}

static __device__ inline unsigned f2mono(float f) {
  unsigned u = __float_as_uint(f);
  return (u & 0x80000000u) ? ~u : (u | 0x80000000u);
}

// A-fragment (16x32, 16-bit): lanes 0-15 K={0..7,16..23}, lanes 16-31 K={8..15,24..31}
static __device__ inline v16bf load_fragA_bf16(const __bf16* rowp, int k0, int kb) {
  Frag16 f;
  f.q[0] = *(const v4u*)(rowp + k0 + kb);
  f.q[1] = *(const v4u*)(rowp + k0 + 16 + kb);
  return f.v;
}

// B-fragment (32x16, 16-bit): lanes 0-15 K={0..7,16..23}, lanes 16-31 K={8..15,24..31}.
// rowp points at row n of B^T (row-major), i.e. column n of B.
static __device__ inline v16bf load_fragB_bf16(const __bf16* rowp, int k0, int kh) {
  Frag16 f;
  f.q[0] = *(const v4u*)(rowp + k0 + kh);
  f.q[1] = *(const v4u*)(rowp + k0 + kh + 16);
  return f.v;
}

// A-fragment from f32 source with bf16 hi/lo split
static __device__ inline void load_fragA_f32_split(const float* rowp, int k0, int kb,
                                                   v16bf* hi, v16bf* lo) {
  Frag16 h, l;
#pragma unroll
  for (int half = 0; half < 2; ++half) {
    v4f x0 = *(const v4f*)(rowp + k0 + half * 16 + kb);
    v4f x1 = *(const v4f*)(rowp + k0 + half * 16 + kb + 4);
#pragma unroll
    for (int e = 0; e < 4; ++e) {
      float fv = x0[e];
      __bf16 hb = (__bf16)fv;
      h.v[half * 8 + e] = hb;
      l.v[half * 8 + e] = (__bf16)(fv - (float)hb);
      fv = x1[e];
      hb = (__bf16)fv;
      h.v[half * 8 + 4 + e] = hb;
      l.v[half * 8 + 4 + e] = (__bf16)(fv - (float)hb);
    }
  }
  *hi = h.v;
  *lo = l.v;
}

// ---------------------------------------------------------------------------
// Transpose + bf16-split a 256x256 weight: Wt[n][k] = W[k][n] (hi/lo pair)
// ---------------------------------------------------------------------------
__global__ __launch_bounds__(256) void prep_wt_kernel(const float* __restrict__ W,
                                                      __bf16* __restrict__ hi,
                                                      __bf16* __restrict__ lo) {
  int idx = blockIdx.x * 256 + threadIdx.x;  // 65536 total
  int n = idx >> 8, k = idx & 255;
  float v = W[k * D_MODEL + n];
  __bf16 h = (__bf16)v;
  hi[idx] = h;
  lo[idx] = (__bf16)(v - (float)h);
}

// ---------------------------------------------------------------------------
// Out(M,256) = A(M,256) @ W(256,256) + bias.  Wt_* hold W transposed, bf16-split.
// writeBf16 != 0 -> write bf16 hi/lo pair buffers (for Q/K), else f32.
// Block: 512 threads = 16 waves; each wave owns a 16-row M-tile.
// ---------------------------------------------------------------------------
__global__ __launch_bounds__(512) void proj_gemm_kernel(
    const float* __restrict__ A, const __bf16* __restrict__ Wt_hi,
    const __bf16* __restrict__ Wt_lo, const float* __restrict__ bias,
    float* __restrict__ outF, __bf16* __restrict__ outHi,
    __bf16* __restrict__ outLo, int writeBf16) {
  const int lane = threadIdx.x & 31;
  const int w    = threadIdx.x >> 5;
  const int m0   = blockIdx.x * 256 + w * 16;
  const int kb   = (lane >> 4) * 8;   // A half-wave K offset
  const int kh   = (lane >> 4) * 8;   // B half-wave K offset

  v16bf ahi[8], alo[8];
  const float* rowp = A + (size_t)(m0 + (lane & 15)) * D_MODEL;
#pragma unroll
  for (int c = 0; c < 8; ++c)
    load_fragA_f32_split(rowp, c * 32, kb, &ahi[c], &alo[c]);

#pragma unroll 1
  for (int nt = 0; nt < 16; ++nt) {
    v8f acc = {0.f, 0.f, 0.f, 0.f, 0.f, 0.f, 0.f, 0.f};
    const int ncol = nt * 16 + (lane & 15);
    const __bf16* bh = Wt_hi + (size_t)ncol * D_MODEL;
    const __bf16* bl = Wt_lo + (size_t)ncol * D_MODEL;
#pragma unroll
    for (int c = 0; c < 8; ++c) {
      v16bf bhi = load_fragB_bf16(bh, c * 32, kh);
      v16bf blo = load_fragB_bf16(bl, c * 32, kh);
      acc = wmma_bf16(ahi[c], bhi, acc);
      acc = wmma_bf16(ahi[c], blo, acc);
      acc = wmma_bf16(alo[c], bhi, acc);
    }
    float bv = bias[ncol];
#pragma unroll
    for (int i = 0; i < 8; ++i) {
      float r = acc[i] + bv;
      int row = m0 + (lane >> 4) * 8 + i;
      size_t oidx = (size_t)row * D_MODEL + ncol;
      if (writeBf16) {
        __bf16 hv = (__bf16)r;
        outHi[oidx] = hv;
        outLo[oidx] = (__bf16)(r - (float)hv);
      } else {
        outF[oidx] = r;
      }
    }
  }
}

// ---------------------------------------------------------------------------
// Fused sparse attention: scores (WMMA) -> exact top-409 radix select ->
// softmax -> sparse weighted-V accumulation.
// Block: 512 threads = 16 waves, one 16-query tile per block.
// LDS: scores 16x4096 f32 (256 KB) + per-row 256-bin histograms (16 KB).
// ---------------------------------------------------------------------------
__global__ __launch_bounds__(512) void sparse_attn_kernel(
    const __bf16* __restrict__ Qhi, const __bf16* __restrict__ Qlo,
    const __bf16* __restrict__ Khi, const __bf16* __restrict__ Klo,
    const float* __restrict__ V, float* __restrict__ attnOut) {
  extern __shared__ char smem[];
  float*    S    = (float*)smem;                              // [16][LIN]
  unsigned* hist = (unsigned*)(smem + (size_t)16 * LIN * 4);  // [16][256]

  const int lane  = threadIdx.x & 31;
  const int w     = threadIdx.x >> 5;
  const int n     = blockIdx.x >> 7;   // / (LQ/16)
  const int qbase = (blockIdx.x & 127) * 16;

  // ---- Phase 1: scores tile S[16][4096] via WMMA -------------------------
  const int kb = (lane >> 4) * 8;
  const int kh = (lane >> 4) * 8;
  const int qrow = n * LQ + qbase + (lane & 15);
  v16bf ahi[8], alo[8];
  {
    const __bf16* qh = Qhi + (size_t)qrow * D_MODEL;
    const __bf16* ql = Qlo + (size_t)qrow * D_MODEL;
#pragma unroll
    for (int c = 0; c < 8; ++c) {
      ahi[c] = load_fragA_bf16(qh, c * 32, kb);
      alo[c] = load_fragA_bf16(ql, c * 32, kb);
    }
  }
#pragma unroll 1
  for (int t = 0; t < 16; ++t) {
    int kt = w * 16 + t;  // 16 waves cover all 256 k-tiles
    int krow = n * LIN + kt * 16 + (lane & 15);
    const __bf16* bh = Khi + (size_t)krow * D_MODEL;
    const __bf16* bl = Klo + (size_t)krow * D_MODEL;
    v8f acc = {0.f, 0.f, 0.f, 0.f, 0.f, 0.f, 0.f, 0.f};
#pragma unroll
    for (int c = 0; c < 8; ++c) {
      v16bf bhi = load_fragB_bf16(bh, c * 32, kh);
      v16bf blo = load_fragB_bf16(bl, c * 32, kh);
      acc = wmma_bf16(ahi[c], bhi, acc);
      acc = wmma_bf16(ahi[c], blo, acc);
      acc = wmma_bf16(alo[c], bhi, acc);
    }
    int col = kt * 16 + (lane & 15);
#pragma unroll
    for (int i = 0; i < 8; ++i) {
      int row = (lane >> 4) * 8 + i;
      S[row * LIN + col] = acc[i];
    }
  }
  __syncthreads();

  // ---- Phase 2: per-row (wave w == row r) max + exact top-K threshold ----
  const int r = w;
  float* Sr = S + r * LIN;
  float rowmax = -3.4e38f;
  for (int j = lane; j < LIN; j += 32) rowmax = fmaxf(rowmax, Sr[j]);
#pragma unroll
  for (int m = 16; m >= 1; m >>= 1) rowmax = fmaxf(rowmax, __shfl_xor(rowmax, m, 32));

  unsigned prefix = 0;
  int need = KTOP;
  unsigned* hr = hist + r * 256;
  for (int p = 24; p >= 0; p -= 8) {
    for (int b = lane; b < 256; b += 32) hr[b] = 0u;
    __syncthreads();
    unsigned mask = (p == 24) ? 0u : (0xFFFFFFFFu << (p + 8));
    for (int j = lane; j < LIN; j += 32) {
      unsigned u = f2mono(Sr[j]);
      if ((u & mask) == (prefix & mask)) atomicAdd(&hr[(u >> p) & 255u], 1u);
    }
    __syncthreads();
    int dsel = 0, needNew = need;
    if (lane == 0) {
      int accn = 0, d = 255;
      for (; d > 0; --d) {
        int cbin = (int)hr[d];
        if (accn + cbin >= need) break;
        accn += cbin;
      }
      dsel = d;
      needNew = need - accn;
    }
    dsel = __shfl(dsel, 0, 32);
    need = __shfl(needNew, 0, 32);
    prefix |= ((unsigned)dsel) << p;
  }
  const unsigned uT = prefix;  // mono-encoding of the 409th-largest score

  float zsum = 0.f;
  for (int j = lane; j < LIN; j += 32) {
    float s = Sr[j];
    if (f2mono(s) >= uT) zsum += __expf(s - rowmax);
  }
#pragma unroll
  for (int m = 16; m >= 1; m >>= 1) zsum += __shfl_xor(zsum, m, 32);
  float invZ = 1.0f / zsum;

  // ---- Phase 3: sparse weighted sum of V rows ----------------------------
  float acc[8];
#pragma unroll
  for (int i = 0; i < 8; ++i) acc[i] = 0.f;
  const float* Vb = V + (size_t)n * LIN * D_MODEL;
  for (int jb = 0; jb < LIN; jb += 32) {
    float s = Sr[jb + lane];
    float wgt = (f2mono(s) >= uT) ? (__expf(s - rowmax) * invZ) : 0.f;
#pragma unroll 1
    for (int src = 0; src < 32; ++src) {
      float wj = __shfl(wgt, src, 32);
      if (wj != 0.f) {  // wave-uniform branch
        const v4f* vp = (const v4f*)(Vb + (size_t)(jb + src) * D_MODEL + lane * 8);
        v4f v0 = vp[0], v1 = vp[1];
#pragma unroll
        for (int e = 0; e < 4; ++e) {
          acc[e]     += wj * v0[e];
          acc[4 + e] += wj * v1[e];
        }
      }
    }
  }

  // ---- Phase 4: write attention output row -------------------------------
  float* op = attnOut + (size_t)(n * LQ + qbase + r) * D_MODEL + lane * 8;
  v4f o0, o1;
#pragma unroll
  for (int e = 0; e < 4; ++e) { o0[e] = acc[e]; o1[e] = acc[4 + e]; }
  *(v4f*)op       = o0;
  *((v4f*)op + 1) = o1;
}

// ---------------------------------------------------------------------------
extern "C" void kernel_launch(void* const* d_in, const int* in_sizes, int n_in,
                              void* d_out, int out_size, void* d_ws, size_t ws_size,
                              hipStream_t stream) {
  (void)in_sizes; (void)n_in; (void)out_size; (void)ws_size;
  const float* query         = (const float*)d_in[0];
  const float* input_flatten = (const float*)d_in[1];
  const float* Wq = (const float*)d_in[2];
  const float* bq = (const float*)d_in[3];
  const float* Wk = (const float*)d_in[4];
  const float* bk = (const float*)d_in[5];
  const float* Wo = (const float*)d_in[6];
  const float* bo = (const float*)d_in[7];
  float* out = (float*)d_out;

  char* ws = (char*)d_ws;
  size_t off = 0;
  auto alloc = [&](size_t bytes) -> void* {
    void* p = ws + off;
    off += bytes;
    off = (off + 255) & ~(size_t)255;
    return p;
  };

  const size_t wtN = (size_t)D_MODEL * D_MODEL;       // 65536
  __bf16* Wtq_hi = (__bf16*)alloc(wtN * 2);
  __bf16* Wtq_lo = (__bf16*)alloc(wtN * 2);
  __bf16* Wtk_hi = (__bf16*)alloc(wtN * 2);
  __bf16* Wtk_lo = (__bf16*)alloc(wtN * 2);
  __bf16* Wto_hi = (__bf16*)alloc(wtN * 2);
  __bf16* Wto_lo = (__bf16*)alloc(wtN * 2);
  const size_t qN = (size_t)N_BATCH * LQ * D_MODEL;   // 2M
  const size_t kN = (size_t)N_BATCH * LIN * D_MODEL;  // 4M
  __bf16* Qhi = (__bf16*)alloc(qN * 2);
  __bf16* Qlo = (__bf16*)alloc(qN * 2);
  __bf16* Khi = (__bf16*)alloc(kN * 2);
  __bf16* Klo = (__bf16*)alloc(kN * 2);
  float* attnOut = (float*)alloc(qN * 4);

  // 1) weight transpose + bf16 split
  prep_wt_kernel<<<256, 256, 0, stream>>>(Wq, Wtq_hi, Wtq_lo);
  prep_wt_kernel<<<256, 256, 0, stream>>>(Wk, Wtk_hi, Wtk_lo);
  prep_wt_kernel<<<256, 256, 0, stream>>>(Wo, Wto_hi, Wto_lo);

  // 2) Q / K projections (WMMA), outputs stored bf16 hi/lo
  proj_gemm_kernel<<<(N_BATCH * LQ) / 256, 512, 0, stream>>>(
      query, Wtq_hi, Wtq_lo, bq, nullptr, Qhi, Qlo, 1);
  proj_gemm_kernel<<<(N_BATCH * LIN) / 256, 512, 0, stream>>>(
      input_flatten, Wtk_hi, Wtk_lo, bk, nullptr, Khi, Klo, 1);

  // 3) fused sparse attention (scores WMMA + exact top-409 + softmax + V)
  size_t smem = (size_t)16 * LIN * 4 + (size_t)16 * 256 * 4;  // 278528 B
  (void)hipFuncSetAttribute((const void*)sparse_attn_kernel,
                            hipFuncAttributeMaxDynamicSharedMemorySize, (int)smem);
  sparse_attn_kernel<<<N_BATCH * (LQ / 16), 512, smem, stream>>>(
      Qhi, Qlo, Khi, Klo, input_flatten, attnOut);

  // 4) output projection -> d_out (f32)
  proj_gemm_kernel<<<(N_BATCH * LQ) / 256, 512, 0, stream>>>(
      attnOut, Wto_hi, Wto_lo, bo, out, nullptr, nullptr, 0);
}
